// CVFSMN_72962904424831
// MI455X (gfx1250) — matrix operand order repaired and checked by reading, not compiled, not run on hardware
//
#include <hip/hip_runtime.h>

typedef __attribute__((ext_vector_type(8)))  float   v8f;
typedef __attribute__((ext_vector_type(8)))  __bf16  v8bf;
typedef __attribute__((ext_vector_type(16))) __bf16  v16bf;

#define BATCH   8
#define TSTEPS  2048
#define DDIM    1024
#define MROWS   (BATCH * TSTEPS)   // 16384
#define MEMN    40

// ---------------- GEMM: C[M][N] = A[M][K] @ W[K][N] + bias, fp32 I/O, split-bf16 WMMA ----
#define BM 128
#define BN 64
#define BK 32
#define LDS_STRIDE 40   // halves per row, 80B: keeps 16B alignment, staggers banks

__global__ __launch_bounds__(256) void cvfsmn_gemm_bf16x3(
    const float* __restrict__ A, const float* __restrict__ W,
    const float* __restrict__ bias, float* __restrict__ C,
    int Mtot, int N, int K)
{
  __shared__ __bf16 As_hi[BM][LDS_STRIDE];
  __shared__ __bf16 As_lo[BM][LDS_STRIDE];
  __shared__ __bf16 Bs_hi[BN][LDS_STRIDE];   // stored transposed: [n][k]
  __shared__ __bf16 Bs_lo[BN][LDS_STRIDE];

  const int tid   = threadIdx.x;
  const int lane  = tid & 31;        // wave32
  const int wid   = tid >> 5;        // 0..7
  const int waveM = wid >> 1;        // 0..3
  const int waveN = wid & 1;         // 0..1
  const int half  = lane >> 4;       // lane half (0/1)
  const int l16   = lane & 15;

  const int row0 = blockIdx.y * BM;
  const int col0 = blockIdx.x * BN;

  const v8f z8 = {0.f, 0.f, 0.f, 0.f, 0.f, 0.f, 0.f, 0.f};
  v8f acc[2][2] = {{z8, z8}, {z8, z8}};

  for (int k0 = 0; k0 < K; k0 += BK) {
    // ---- stage A tile (128 x 32 fp32 -> bf16 hi/lo), 256 thr x 4 float4
#pragma unroll
    for (int i = 0; i < 4; ++i) {
      int lin = tid + i * 256;
      int r = lin >> 3;
      int c = (lin & 7) << 2;
      const float4 v = *reinterpret_cast<const float4*>(
          &A[(size_t)(row0 + r) * K + k0 + c]);
      float vx[4] = {v.x, v.y, v.z, v.w};
#pragma unroll
      for (int j = 0; j < 4; ++j) {
        __bf16 h = (__bf16)vx[j];
        As_hi[r][c + j] = h;
        As_lo[r][c + j] = (__bf16)(vx[j] - (float)h);
      }
    }
    // ---- stage B tile (32 x 64 fp32 -> bf16 hi/lo, transposed to [n][k])
#pragma unroll
    for (int i = 0; i < 2; ++i) {
      int lin = tid + i * 256;
      int kr = lin >> 4;
      int nc = (lin & 15) << 2;
      const float4 v = *reinterpret_cast<const float4*>(
          &W[(size_t)(k0 + kr) * N + col0 + nc]);
      float vx[4] = {v.x, v.y, v.z, v.w};
#pragma unroll
      for (int j = 0; j < 4; ++j) {
        __bf16 h = (__bf16)vx[j];
        Bs_hi[nc + j][kr] = h;
        Bs_lo[nc + j][kr] = (__bf16)(vx[j] - (float)h);
      }
    }
    __syncthreads();

    // ---- operand fetch per ISA VGPR layouts
    v16bf a_hi[2], a_lo[2], b_hi[2], b_lo[2];
#pragma unroll
    for (int tm = 0; tm < 2; ++tm) {
      int r = waveM * 32 + tm * 16 + l16;
      // A 16x32 bf16: lane half 0 -> K {0..7, 16..23}; half 1 -> K {8..15, 24..31}
      v8bf x0 = *reinterpret_cast<const v8bf*>(&As_hi[r][half * 8]);
      v8bf x1 = *reinterpret_cast<const v8bf*>(&As_hi[r][half * 8 + 16]);
      a_hi[tm] = __builtin_shufflevector(x0, x1, 0,1,2,3,4,5,6,7,8,9,10,11,12,13,14,15);
      v8bf y0 = *reinterpret_cast<const v8bf*>(&As_lo[r][half * 8]);
      v8bf y1 = *reinterpret_cast<const v8bf*>(&As_lo[r][half * 8 + 16]);
      a_lo[tm] = __builtin_shufflevector(y0, y1, 0,1,2,3,4,5,6,7,8,9,10,11,12,13,14,15);
    }
#pragma unroll
    for (int tn = 0; tn < 2; ++tn) {
      int n = waveN * 32 + tn * 16 + l16;
      // B 32x16 bf16: lane half 0 -> K {0..7, 16..23}; half 1 -> K {8..15, 24..31}
      v8bf x0 = *reinterpret_cast<const v8bf*>(&Bs_hi[n][half * 8]);
      v8bf x1 = *reinterpret_cast<const v8bf*>(&Bs_hi[n][half * 8 + 16]);
      b_hi[tn] = __builtin_shufflevector(x0, x1, 0,1,2,3,4,5,6,7,8,9,10,11,12,13,14,15);
      v8bf y0 = *reinterpret_cast<const v8bf*>(&Bs_lo[n][half * 8]);
      v8bf y1 = *reinterpret_cast<const v8bf*>(&Bs_lo[n][half * 8 + 16]);
      b_lo[tn] = __builtin_shufflevector(y0, y1, 0,1,2,3,4,5,6,7,8,9,10,11,12,13,14,15);
    }

#pragma unroll
    for (int tm = 0; tm < 2; ++tm)
#pragma unroll
      for (int tn = 0; tn < 2; ++tn) {
        acc[tm][tn] = __builtin_amdgcn_wmma_f32_16x16x32_bf16(
            false, a_hi[tm], false, b_hi[tn], (short)0, acc[tm][tn], false, false);
        acc[tm][tn] = __builtin_amdgcn_wmma_f32_16x16x32_bf16(
            false, a_lo[tm], false, b_hi[tn], (short)0, acc[tm][tn], false, false);
        acc[tm][tn] = __builtin_amdgcn_wmma_f32_16x16x32_bf16(
            false, a_hi[tm], false, b_lo[tn], (short)0, acc[tm][tn], false, false);
      }
    __syncthreads();
  }

  // ---- epilogue: C/D layout: VGPR r -> M = half*8 + r, N = l16
#pragma unroll
  for (int tn = 0; tn < 2; ++tn) {
    int colG = col0 + waveN * 32 + tn * 16 + l16;
    float bv = bias[colG];
#pragma unroll
    for (int tm = 0; tm < 2; ++tm) {
#pragma unroll
      for (int r8 = 0; r8 < 8; ++r8) {
        int rowG = row0 + waveM * 32 + tm * 16 + half * 8 + r8;
        C[(size_t)rowG * N + colG] = acc[tm][tn][r8] + bv;
      }
    }
  }
}

// ---------------- totals[b,d] = sum_t P[b,t,d] ----------------
__global__ void cvfsmn_zero(float* __restrict__ p, int n) {
  int i = blockIdx.x * blockDim.x + threadIdx.x;
  if (i < n) p[i] = 0.f;
}

__global__ void cvfsmn_reduce_t(const float* __restrict__ P, float* __restrict__ totals) {
  int d  = blockIdx.x * 256 + threadIdx.x;
  int t0 = blockIdx.y * 256;
  int b  = blockIdx.z;
  float s = 0.f;
  const float* base = P + ((size_t)b * TSTEPS + t0) * DDIM + d;
  for (int t = 0; t < 256; ++t) s += base[(size_t)t * DDIM];
  atomicAdd(&totals[b * DDIM + d], s);
}

// ---------------- FSMN memory: band + mw0*(total - window_sum) ----------------
#define TT 16
#define HALO (MEMN - 1)        // 39
#define ROWS (TT + MEMN - 1)   // 55

__global__ __launch_bounds__(256) void cvfsmn_memory(
    const float* __restrict__ P, const float* __restrict__ mw,
    const float* __restrict__ totals, float* __restrict__ Pm)
{
  __shared__ float tile[ROWS][256];   // 56320 B

  const int d  = blockIdx.y * 256 + threadIdx.x;
  const int t0 = blockIdx.x * TT;
  const int b  = blockIdx.z;

  // stage P[t0-39 .. t0+15] for this channel slab (zeros before t=0)
  for (int r = 0; r < ROWS; ++r) {
    int tg = t0 - HALO + r;
    tile[r][threadIdx.x] =
        (tg >= 0) ? P[((size_t)b * TSTEPS + tg) * DDIM + d] : 0.f;
  }
  __syncthreads();

  float acc[TT], wsum[TT];
#pragma unroll
  for (int t = 0; t < TT; ++t) { acc[t] = 0.f; wsum[t] = 0.f; }

  for (int k = 0; k < MEMN; ++k) {
    float w = mw[(size_t)(k + 1) * DDIM + d];
#pragma unroll
    for (int t = 0; t < TT; ++t) {
      float v = tile[HALO - k + t][threadIdx.x];
      acc[t]  = fmaf(w, v, acc[t]);
      wsum[t] += v;
    }
  }

  const float w0  = mw[d];
  const float tot = totals[b * DDIM + d];
#pragma unroll
  for (int t = 0; t < TT; ++t) {
    Pm[((size_t)b * TSTEPS + t0 + t) * DDIM + d] =
        acc[t] + w0 * (tot - wsum[t]);
  }
}

// ---------------- launcher ----------------
extern "C" void kernel_launch(void* const* d_in, const int* in_sizes, int n_in,
                              void* d_out, int out_size, void* d_ws, size_t ws_size,
                              hipStream_t stream) {
  const float* x     = (const float*)d_in[0];   // (8, 2048, 1024)
  const float* W1    = (const float*)d_in[1];   // (1024, 1024)
  const float* bias1 = (const float*)d_in[2];   // (1024,)
  const float* W2    = (const float*)d_in[3];   // (1024, 1024)
  const float* bias2 = (const float*)d_in[4];   // (1024,)
  const float* mw    = (const float*)d_in[5];   // (41, 1024)
  float* out = (float*)d_out;

  float* P      = (float*)d_ws;                    // 64 MB
  float* Pm     = P  + (size_t)MROWS * DDIM;       // 64 MB
  float* totals = Pm + (size_t)MROWS * DDIM;       // 32 KB

  dim3 gGemm(DDIM / BN, MROWS / BM);               // 16 x 128

  // 1) P = X @ W1 + b1
  cvfsmn_gemm_bf16x3<<<gGemm, 256, 0, stream>>>(x, W1, bias1, P, MROWS, DDIM, DDIM);

  // 2) totals[b,d] = sum_t P
  cvfsmn_zero<<<(BATCH * DDIM + 255) / 256, 256, 0, stream>>>(totals, BATCH * DDIM);
  cvfsmn_reduce_t<<<dim3(DDIM / 256, TSTEPS / 256, BATCH), 256, 0, stream>>>(P, totals);

  // 3) FSMN memory
  cvfsmn_memory<<<dim3(TSTEPS / TT, DDIM / 256, BATCH), 256, 0, stream>>>(P, mw, totals, Pm);

  // 4) out = Pm @ W2 + b2
  cvfsmn_gemm_bf16x3<<<gGemm, 256, 0, stream>>>(Pm, W2, bias2, out, MROWS, DDIM, DDIM);
}
